// Attention_5686536699939
// MI455X (gfx1250) — matrix operand order repaired and checked by reading, not compiled, not run on hardware
//
#include <hip/hip_runtime.h>
#include <hip/hip_bf16.h>
#include <stdint.h>

typedef __attribute__((ext_vector_type(16))) __bf16 v16bf;
typedef __attribute__((ext_vector_type(8)))  __bf16 v8bf;
typedef __attribute__((ext_vector_type(8)))  float  v8f;

#define NTOK   197
#define NP     224          // token dim padded to 7*32 for WMMA K-steps
#define NH     12
#define HD     64
#define DIM    768
#define BATCH  64
#define MROWS  (BATCH * NTOK)   // 12608 = 788*16 exactly
#define MTILES (MROWS / 16)     // 788
#define RPS    208              // padded rel-pos-bias tile (13*16)

// ---- WMMA helpers (CDNA5 bf16 16x16x32, f32 accum) -------------------------
static __device__ __forceinline__ v8f wmma_bf16(v16bf a, v16bf b, v8f c) {
  return __builtin_amdgcn_wmma_f32_16x16x32_bf16(false, a, false, b,
                                                 (short)0, c, false, false);
}

// A-fragment: lane l holds row (l&15); element e -> K = kstep*32 + (l>>4)*16 + e
// caller passes k0 = kstep*32 + (l>>4)*8
static __device__ __forceinline__ v16bf load_afrag(const __bf16* row, int k0) {
  v8bf lo = *(const v8bf*)(row + k0 + (k0 & 8));
  v8bf hi = *(const v8bf*)(row + k0 + (k0 & 8) + 8);
  v16bf a;
#pragma unroll
  for (int e = 0; e < 8; ++e) { a[e] = lo[e]; a[e + 8] = hi[e]; }
  return a;
}

// ---- small utility kernels -------------------------------------------------
__global__ void f2bf_kernel(const float* __restrict__ s, __bf16* __restrict__ d, int n) {
  for (int i = blockIdx.x * blockDim.x + threadIdx.x; i < n; i += gridDim.x * blockDim.x)
    d[i] = (__bf16)s[i];
}

__global__ void zero16_kernel(uint4* __restrict__ p, size_t n16) {
  uint4 z = make_uint4(0u, 0u, 0u, 0u);
  for (size_t i = blockIdx.x * (size_t)blockDim.x + threadIdx.x; i < n16;
       i += (size_t)gridDim.x * blockDim.x)
    p[i] = z;
}

// rpb_full[h][i][j] = rpb_table[rel_idx[i,j], h], zero-padded to 208x208
__global__ void rpb_gather_kernel(const float* __restrict__ table,
                                  const int* __restrict__ idx,
                                  float* __restrict__ out) {
  const int total = NH * RPS * RPS;
  for (int t = blockIdx.x * blockDim.x + threadIdx.x; t < total;
       t += gridDim.x * blockDim.x) {
    int h = t / (RPS * RPS);
    int rem = t % (RPS * RPS);
    int i = rem / RPS, j = rem % RPS;
    float v = 0.f;
    if (i < NTOK && j < NTOK) v = table[idx[i * NTOK + j] * NH + h];
    out[t] = v;
  }
}

// ---- QKV GEMM: y = x @ qkv_w^T + [q_bias,0,v_bias]; scatter q,k,v ---------
// one wave -> 16x64 output tile. M = 12608, K = 768, N = 2304.
__global__ __launch_bounds__(128) void qkv_gemm_kernel(
    const __bf16* __restrict__ xb,   // [12608][768]
    const __bf16* __restrict__ wb,   // [2304][768] (row-major, K contiguous)
    const float* __restrict__ qbias,
    const float* __restrict__ vbias,
    __bf16* __restrict__ qs,         // [B][H][NP][HD] (scaled q)
    __bf16* __restrict__ ksb,        // [B][H][NP][HD]
    __bf16* __restrict__ vtb)        // [B][H][HD][NP] (transposed v)
{
  const int NT64 = 36;  // 2304 / 64
  int wave = blockIdx.x * (blockDim.x >> 5) + (threadIdx.x >> 5);
  if (wave >= MTILES * NT64) return;
  int lane = threadIdx.x & 31, lhalf = lane >> 4, l16 = lane & 15;
  int mtile = wave / NT64;
  int n64 = (wave % NT64) * 64;

  const __bf16* arow = xb + (size_t)(mtile * 16 + l16) * DIM;
  const __bf16* b0 = wb + (size_t)(n64 + l16) * DIM + lhalf * 16;

  v8f acc[4] = {};
#pragma unroll 4
  for (int kk = 0; kk < 24; ++kk) {
    if (kk + 8 < 24) __builtin_prefetch(arow + (kk + 8) * 32, 0, 3);
    v16bf a = load_afrag(arow, kk * 32 + lhalf * 8);
    v16bf bf0 = *(const v16bf*)(b0 + kk * 32);
    v16bf bf1 = *(const v16bf*)(b0 + 16 * DIM + kk * 32);
    v16bf bf2 = *(const v16bf*)(b0 + 32 * DIM + kk * 32);
    v16bf bf3 = *(const v16bf*)(b0 + 48 * DIM + kk * 32);
    acc[0] = wmma_bf16(a, bf0, acc[0]);
    acc[1] = wmma_bf16(a, bf1, acc[1]);
    acc[2] = wmma_bf16(a, bf2, acc[2]);
    acc[3] = wmma_bf16(a, bf3, acc[3]);
  }

  int bi[8], tk[8];
#pragma unroll
  for (int r = 0; r < 8; ++r) {
    int M = mtile * 16 + r + 8 * lhalf;
    bi[r] = M / NTOK;
    tk[r] = M % NTOK;
  }

#pragma unroll
  for (int t = 0; t < 4; ++t) {
    int n = n64 + t * 16 + l16;
    int h = (n % DIM) / HD, d = n % HD;
    if (n < DIM) {                       // q: scale by 1/sqrt(64)
      float bias = qbias[n];
#pragma unroll
      for (int r = 0; r < 8; ++r) {
        float v = (acc[t][r] + bias) * 0.125f;
        qs[(((size_t)bi[r] * NH + h) * NP + tk[r]) * HD + d] = (__bf16)v;
      }
    } else if (n < 2 * DIM) {            // k: zero bias
#pragma unroll
      for (int r = 0; r < 8; ++r)
        ksb[(((size_t)bi[r] * NH + h) * NP + tk[r]) * HD + d] = (__bf16)acc[t][r];
    } else {                             // v: transposed store
      float bias = vbias[n - 2 * DIM];
#pragma unroll
      for (int r = 0; r < 8; ++r) {
        float v = acc[t][r] + bias;
        vtb[(((size_t)bi[r] * NH + h) * HD + d) * NP + tk[r]] = (__bf16)v;
      }
    }
  }
}

// ---- fused attention: scores + rpb + softmax + PV, one wave per row-tile ---
__global__ __launch_bounds__(32) void attn_kernel(
    const __bf16* __restrict__ qs, const __bf16* __restrict__ ksb,
    const __bf16* __restrict__ vtb, const float* __restrict__ rpbf,
    __bf16* __restrict__ ao)  // [B*197][768] bf16
{
  __shared__ uint4 Pu[16 * NP * 2 / 16];  // 7168 B probability tile
  __bf16* P = (__bf16*)Pu;

  int blk = blockIdx.x;
  int mtile = blk % 13;
  int bh = blk / 13;
  int h = bh % NH;
  int b = bh / NH;
  int lane = threadIdx.x, lhalf = lane >> 4, l16 = lane & 15;

  const __bf16* qrow  = qs  + ((size_t)bh * NP + mtile * 16 + l16) * HD;
  const __bf16* krows = ksb + (size_t)bh * NP * HD;
  const __bf16* vrows = vtb + (size_t)bh * HD * NP;

  // Q fragments for K=0..31 and K=32..63
  v16bf aq0 = load_afrag(qrow, lhalf * 8);
  v16bf aq1 = load_afrag(qrow, 32 + lhalf * 8);

  // scores: 13 column tiles of 16, kept in VGPRs
  v8f s[13];
  const float* rpb_base = rpbf + ((size_t)h * RPS + mtile * 16 + 8 * lhalf) * RPS;
#pragma unroll
  for (int t = 0; t < 13; ++t) {
    const __bf16* kr = krows + (size_t)(t * 16 + l16) * HD + lhalf * 16;
    v16bf bk0 = *(const v16bf*)kr;
    v16bf bk1 = *(const v16bf*)(kr + 32);
    v8f c = {};
    c = wmma_bf16(aq0, bk0, c);
    c = wmma_bf16(aq1, bk1, c);
    int j = t * 16 + l16;
    if (j < NTOK) {
      const float* rp = rpb_base + j;
#pragma unroll
      for (int r = 0; r < 8; ++r) c[r] += rp[(size_t)r * RPS];
    } else {
#pragma unroll
      for (int r = 0; r < 8; ++r) c[r] = -1e30f;  // mask padded columns
    }
    s[t] = c;
  }

  // row-wise softmax: row (r + 8*(l>>4)) lives in one 16-lane half
  float inv[8];
#pragma unroll
  for (int r = 0; r < 8; ++r) {
    float m = s[0][r];
#pragma unroll
    for (int t = 1; t < 13; ++t) m = fmaxf(m, s[t][r]);
    m = fmaxf(m, __shfl_xor(m, 1));
    m = fmaxf(m, __shfl_xor(m, 2));
    m = fmaxf(m, __shfl_xor(m, 4));
    m = fmaxf(m, __shfl_xor(m, 8));
    float sum = 0.f;
#pragma unroll
    for (int t = 0; t < 13; ++t) {
      float e = __expf(s[t][r] - m);
      s[t][r] = e;
      sum += e;
    }
    sum += __shfl_xor(sum, 1);
    sum += __shfl_xor(sum, 2);
    sum += __shfl_xor(sum, 4);
    sum += __shfl_xor(sum, 8);
    inv[r] = 1.0f / sum;
  }

  // re-layout P (C layout -> A layout) through LDS; zero K-padding 208..223
  uint4 z = make_uint4(0u, 0u, 0u, 0u);
#pragma unroll
  for (int i = lane; i < 16 * NP * 2 / 16; i += 32) Pu[i] = z;
  __syncthreads();
#pragma unroll
  for (int t = 0; t < 13; ++t) {
    int col = t * 16 + l16;
#pragma unroll
    for (int r = 0; r < 8; ++r)
      P[(r + 8 * lhalf) * NP + col] = (__bf16)(s[t][r] * inv[r]);
  }
  __syncthreads();

  // PV: out[16][64] = P[16][224] @ V[224][64], 7 K-steps of 32
  v16bf ap[7];
  const __bf16* prow = P + l16 * NP;
#pragma unroll
  for (int k2 = 0; k2 < 7; ++k2) ap[k2] = load_afrag(prow, k2 * 32 + lhalf * 8);

#pragma unroll
  for (int dt = 0; dt < 4; ++dt) {
    v8f acc = {};
    const __bf16* vr = vrows + (size_t)(dt * 16 + l16) * NP + lhalf * 16;
#pragma unroll
    for (int k2 = 0; k2 < 7; ++k2) {
      v16bf bv = *(const v16bf*)(vr + k2 * 32);
      acc = wmma_bf16(ap[k2], bv, acc);
    }
#pragma unroll
    for (int r = 0; r < 8; ++r) {
      int tok = mtile * 16 + r + 8 * lhalf;
      if (tok < NTOK)
        ao[((size_t)b * NTOK + tok) * DIM + h * HD + dt * 16 + l16] = (__bf16)acc[r];
    }
  }
}

// ---- projection GEMM: out = ao @ proj_w^T + proj_b (fp32 output) -----------
__global__ __launch_bounds__(128) void proj_gemm_kernel(
    const __bf16* __restrict__ ab,   // [12608][768]
    const __bf16* __restrict__ wb,   // [768][768]
    const float* __restrict__ pbias,
    float* __restrict__ out)         // [12608][768]
{
  const int NT64 = 12;  // 768 / 64
  int wave = blockIdx.x * (blockDim.x >> 5) + (threadIdx.x >> 5);
  if (wave >= MTILES * NT64) return;
  int lane = threadIdx.x & 31, lhalf = lane >> 4, l16 = lane & 15;
  int mtile = wave / NT64;
  int n64 = (wave % NT64) * 64;

  const __bf16* arow = ab + (size_t)(mtile * 16 + l16) * DIM;
  const __bf16* b0 = wb + (size_t)(n64 + l16) * DIM + lhalf * 16;

  v8f acc[4] = {};
#pragma unroll 4
  for (int kk = 0; kk < 24; ++kk) {
    if (kk + 8 < 24) __builtin_prefetch(arow + (kk + 8) * 32, 0, 3);
    v16bf a = load_afrag(arow, kk * 32 + lhalf * 8);
    v16bf bf0 = *(const v16bf*)(b0 + kk * 32);
    v16bf bf1 = *(const v16bf*)(b0 + 16 * DIM + kk * 32);
    v16bf bf2 = *(const v16bf*)(b0 + 32 * DIM + kk * 32);
    v16bf bf3 = *(const v16bf*)(b0 + 48 * DIM + kk * 32);
    acc[0] = wmma_bf16(a, bf0, acc[0]);
    acc[1] = wmma_bf16(a, bf1, acc[1]);
    acc[2] = wmma_bf16(a, bf2, acc[2]);
    acc[3] = wmma_bf16(a, bf3, acc[3]);
  }

#pragma unroll
  for (int t = 0; t < 4; ++t) {
    int n = n64 + t * 16 + l16;
    float bias = pbias[n];
#pragma unroll
    for (int r = 0; r < 8; ++r) {
      int M = mtile * 16 + r + 8 * lhalf;
      out[(size_t)M * DIM + n] = acc[t][r] + bias;
    }
  }
}

// ---- host launch -----------------------------------------------------------
extern "C" void kernel_launch(void* const* d_in, const int* in_sizes, int n_in,
                              void* d_out, int out_size, void* d_ws, size_t ws_size,
                              hipStream_t stream) {
  (void)in_sizes; (void)n_in; (void)out_size; (void)ws_size;
  const float* x      = (const float*)d_in[0];
  const float* qkv_w  = (const float*)d_in[1];
  const float* q_bias = (const float*)d_in[2];
  const float* v_bias = (const float*)d_in[3];
  const float* rpb_t  = (const float*)d_in[4];
  const float* proj_w = (const float*)d_in[5];
  const float* proj_b = (const float*)d_in[6];
  const int*   relidx = (const int*)d_in[7];

  char* ws = (char*)d_ws;
  size_t off = 0;
  auto alloc = [&](size_t bytes) -> void* {
    void* p = ws + off;
    off += (bytes + 255) & ~(size_t)255;
    return p;
  };
  const size_t qkv_buf = (size_t)BATCH * NH * NP * HD * 2;  // 22,020,096 B each
  __bf16* xb    = (__bf16*)alloc((size_t)MROWS * DIM * 2);
  __bf16* wqkv  = (__bf16*)alloc((size_t)3 * DIM * DIM * 2);
  __bf16* wproj = (__bf16*)alloc((size_t)DIM * DIM * 2);
  __bf16* qs    = (__bf16*)alloc(qkv_buf);
  __bf16* ksb   = (__bf16*)alloc(qkv_buf);
  __bf16* vtb   = (__bf16*)alloc(qkv_buf);
  float*  rpbf  = (float*) alloc((size_t)NH * RPS * RPS * 4);
  __bf16* ao    = (__bf16*)alloc((size_t)MROWS * DIM * 2);

  // stage: fp32 -> bf16 once; zero padded q/k/v staging (contiguous region)
  f2bf_kernel<<<2048, 256, 0, stream>>>(x, xb, MROWS * DIM);
  f2bf_kernel<<<1024, 256, 0, stream>>>(qkv_w, wqkv, 3 * DIM * DIM);
  f2bf_kernel<<<512, 256, 0, stream>>>(proj_w, wproj, DIM * DIM);
  zero16_kernel<<<2048, 256, 0, stream>>>((uint4*)qs, (3 * qkv_buf) / 16);
  rpb_gather_kernel<<<1024, 256, 0, stream>>>(rpb_t, relidx, rpbf);

  qkv_gemm_kernel<<<(MTILES * 36) / 4, 128, 0, stream>>>(
      xb, wqkv, q_bias, v_bias, qs, ksb, vtb);
  attn_kernel<<<BATCH * NH * 13, 32, 0, stream>>>(qs, ksb, vtb, rpbf, ao);
  proj_gemm_kernel<<<(MTILES * 12) / 4, 128, 0, stream>>>(
      ao, wproj, proj_b, (float*)d_out);
}
